// GraphAttentionLayer_46351287059136
// MI455X (gfx1250) — matrix operand order repaired and checked by reading, not compiled, not run on hardware
//
#include <hip/hip_runtime.h>
#include <hip/hip_bf16.h>

// Problem constants
constexpr int BB = 8;      // batch
constexpr int NN = 2048;   // nodes
constexpr int FF = 64;     // features (in == out)
constexpr float ALPHA = 0.2f;

typedef __attribute__((ext_vector_type(16))) __bf16 v16bf;
typedef __attribute__((ext_vector_type(8)))  float  v8f;

// ---------------------------------------------------------------------------
// Kernel 1: h = input @ W ; f1 = h@a1 ; f2 = h@a2 ; store h^T as bf16
// One thread per (b, n) row. W (16KB) + a1/a2 staged in LDS.
// h_t layout: [B][F][N] (feature-major) so WMMA B-tiles are contiguous.
// ---------------------------------------------------------------------------
__global__ __launch_bounds__(256) void k1_h_f12(
    const float* __restrict__ input, const float* __restrict__ W,
    const float* __restrict__ a1, const float* __restrict__ a2,
    __bf16* __restrict__ h_t, float* __restrict__ f1, float* __restrict__ f2) {
  __shared__ float sW[FF * FF];
  __shared__ float sa1[FF];
  __shared__ float sa2[FF];
  for (int t = threadIdx.x; t < FF * FF; t += 256) sW[t] = W[t];
  if (threadIdx.x < FF) {
    sa1[threadIdx.x] = a1[threadIdx.x];
    sa2[threadIdx.x] = a2[threadIdx.x];
  }
  __syncthreads();

  const int gid = blockIdx.x * 256 + threadIdx.x;   // 0 .. B*N-1
  const int b = gid / NN;
  const int n = gid % NN;
  const float* in = input + (size_t)gid * FF;

  float h[FF];
#pragma unroll
  for (int o = 0; o < FF; ++o) h[o] = 0.f;
  for (int f = 0; f < FF; ++f) {
    const float x = in[f];
#pragma unroll
    for (int o = 0; o < FF; ++o) h[o] = fmaf(x, sW[f * FF + o], h[o]);
  }

  float s1 = 0.f, s2 = 0.f;
#pragma unroll
  for (int o = 0; o < FF; ++o) {
    s1 = fmaf(h[o], sa1[o], s1);
    s2 = fmaf(h[o], sa2[o], s2);
  }
  f1[gid] = s1;
  f2[gid] = s2;

  __bf16* ht = h_t + (size_t)b * FF * NN + n;
#pragma unroll
  for (int o = 0; o < FF; ++o) ht[(size_t)o * NN] = (__bf16)h[o];
}

// ---------------------------------------------------------------------------
// Kernel 2: partial column sums of exp(adj * leaky_relu(f1_i + f2_j))
// grid: (j-chunk, i-chunk, b). Thread t owns column j = j0+t (coalesced),
// sums 256 rows i. No atomics -> bitwise deterministic.
// ---------------------------------------------------------------------------
__global__ __launch_bounds__(256) void k2_partial(
    const float* __restrict__ adj, const float* __restrict__ f1,
    const float* __restrict__ f2, float* __restrict__ partial) {
  const int b = blockIdx.z;
  const int j = blockIdx.x * 256 + threadIdx.x;
  const int i0 = blockIdx.y * 256;
  const float f2j = f2[b * NN + j];
  const float* arow = adj + (size_t)b * NN * NN + (size_t)i0 * NN + j;
  const float* f1p = f1 + b * NN + i0;

  float acc = 0.f;
  for (int ii = 0; ii < 256; ++ii) {
    const float a = arow[(size_t)ii * NN];
    const float t = f1p[ii] + f2j;
    const float e = t > 0.f ? t : ALPHA * t;
    acc += __expf(a * e);   // adj==0 -> exp(0)=1 (matches reference semantics)
  }
  partial[((size_t)b * 8 + blockIdx.y) * NN + j] = acc;
}

// Kernel 2b: deterministic reduction of the 8 partials -> reciprocal denom
__global__ __launch_bounds__(256) void k2b_rdenom(
    const float* __restrict__ partial, float* __restrict__ rdenom) {
  const int gid = blockIdx.x * 256 + threadIdx.x;   // 0 .. B*N-1
  const int b = gid / NN;
  const int j = gid % NN;
  float s = 0.f;
#pragma unroll
  for (int c = 0; c < 8; ++c) s += partial[((size_t)b * 8 + c) * NN + j];
  rdenom[gid] = __builtin_amdgcn_rcpf(s);
}

// ---------------------------------------------------------------------------
// Kernel 3: out = elu(attention @ h + bias + input)
// One wave per 16(i) x 64(o) tile; K loop over j in steps of 32.
// A tile (attention, bf16) built in registers in contiguous k order (same as B):
//   lanes 0-15:  M=lane,    idx 0..15 -> K=idx
//   lanes 16-31: M=lane-16, idx 0..15 -> K=16+idx
// B tile from h_t[b][o][j] (feature-major bf16): lane l loads 16 contiguous
// bf16 at row o0+(l&15), K-base (l>>4)*16 -> two b128 loads, no transpose.
// ---------------------------------------------------------------------------
__global__ __launch_bounds__(256) void k3_attn_matmul(
    const float* __restrict__ adj, const float* __restrict__ input,
    const float* __restrict__ bias, const __bf16* __restrict__ h_t,
    const float* __restrict__ f1, const float* __restrict__ f2,
    const float* __restrict__ rdenom, float* __restrict__ out) {
  const int wave = threadIdx.x >> 5;
  const int lane = threadIdx.x & 31;
  const int tile = blockIdx.x * 8 + wave;   // 0 .. B*(N/16)-1
  const int b = tile >> 7;                  // tile / (N/16)
  const int i0 = (tile & 127) << 4;         // i-tile base within batch
  const int lm = lane & 15;
  const int hi = lane >> 4;                 // 0 or 1
  const int i = i0 + lm;                    // A-matrix row owned by this lane

  const float f1i = f1[b * NN + i];
  const float* adjrow = adj + (size_t)b * NN * NN + (size_t)i * NN;
  const float* f2b = f2 + b * NN;
  const float* rdb = rdenom + b * NN;
  const __bf16* hb = h_t + (size_t)b * FF * NN;

  const int kbA = hi * 8;    // unused; A is built with kbB (contiguous k)
  const int kbB = hi * 16;   // this lane's K base (B)

  v8f acc[4] = {};           // o-subtiles 0..3 (o0 = 16*s)

  for (int j0 = 0; j0 < NN; j0 += 32) {
    // ---- build attention A tile in registers ----
    v16bf At;
#pragma unroll
    for (int idx = 0; idx < 16; ++idx) {
      const int k = kbB + idx;
      const int j = j0 + k;
      const float a = adjrow[j];
      const float t = f1i + f2b[j];
      const float e = t > 0.f ? t : ALPHA * t;
      const float p = __expf(a * e) * rdb[j];   // softmax over i (axis=1)
      At[idx] = (__bf16)p;
    }
    // ---- 4 WMMAs sharing A, B from feature-major bf16 h ----
#pragma unroll
    for (int s = 0; s < 4; ++s) {
      const v16bf Bt =
          *(const v16bf*)(hb + (size_t)(s * 16 + lm) * NN + j0 + kbB);
      acc[s] = __builtin_amdgcn_wmma_f32_16x16x32_bf16(
          false, At, false, Bt, (short)0, acc[s], false, false);
    }
  }

  // ---- epilogue: + bias + residual, ELU; C/D layout: VGPR r -> M = r + 8*hi
#pragma unroll
  for (int s = 0; s < 4; ++s) {
#pragma unroll
    for (int r = 0; r < 8; ++r) {
      const int row = i0 + hi * 8 + r;
      const int col = s * 16 + lm;
      float x = acc[s][r] + bias[row * FF + col] +
                input[((size_t)b * NN + row) * FF + col];
      const float y = x > 0.f ? x : (__expf(x) - 1.f);
      out[((size_t)b * NN + row) * FF + col] = y;
    }
  }
}

// ---------------------------------------------------------------------------
// Launch: inputs = {input, adj, W, a1, a2, bias}; out f32 [8,2048,64].
// Workspace: h_t bf16 2MB | f1 64KB | f2 64KB | partial 512KB | rdenom 64KB
// ---------------------------------------------------------------------------
extern "C" void kernel_launch(void* const* d_in, const int* in_sizes, int n_in,
                              void* d_out, int out_size, void* d_ws,
                              size_t ws_size, hipStream_t stream) {
  const float* input = (const float*)d_in[0];
  const float* adj   = (const float*)d_in[1];
  const float* W     = (const float*)d_in[2];
  const float* a1    = (const float*)d_in[3];
  const float* a2    = (const float*)d_in[4];
  const float* bias  = (const float*)d_in[5];
  float* out = (float*)d_out;

  char* ws = (char*)d_ws;
  __bf16* h_t   = (__bf16*)ws;                       // B*F*N bf16 = 2 MB
  float* f1     = (float*)(ws + (size_t)(2 << 20));  // B*N
  float* f2     = f1 + BB * NN;                      // B*N
  float* partial= f2 + BB * NN;                      // B*8*N
  float* rdenom = partial + (size_t)BB * 8 * NN;     // B*N

  k1_h_f12<<<BB * NN / 256, 256, 0, stream>>>(input, W, a1, a2, h_t, f1, f2);

  dim3 g2(NN / 256, NN / 256, BB);                   // (8, 8, 8)
  k2_partial<<<g2, 256, 0, stream>>>(adj, f1, f2, partial);
  k2b_rdenom<<<BB * NN / 256, 256, 0, stream>>>(partial, rdenom);

  k3_attn_matmul<<<(BB * NN / 16) / 8, 256, 0, stream>>>(
      adj, input, bias, h_t, f1, f2, rdenom, out);
}
